// PyTorchModel_7945689498039
// MI455X (gfx1250) — matrix operand order repaired and checked by reading, not compiled, not run on hardware
//
#include <hip/hip_runtime.h>
#include <cstdint>

// ---- CDNA5 WMMA types -------------------------------------------------------
typedef __attribute__((ext_vector_type(16))) _Float16 v16h;  // 8 VGPRs
typedef __attribute__((ext_vector_type(8)))  float    v8f;   // 8 VGPRs

union AB16 { v16h h; uint32_t u[8]; };   // A/B operand: elem 2i,2i+1 = VGPR i lo/hi
union C8   { v8f  v; float    f[8]; };   // C/D accumulator

__device__ __forceinline__ uint32_t pkf16(float a, float b) {
  auto h = __builtin_amdgcn_cvt_pkrtz(a, b);     // v_cvt_pk_rtz_f16_f32
  return __builtin_bit_cast(uint32_t, h);
}

// gfx1250 hardware tanh. Round-3 evidence: __builtin_amdgcn_tanhf exists and
// lowers to v_tanh_f32 (single trans op). Keep a ladder for safety.
__device__ __forceinline__ float th(float x) {
#if __has_builtin(__builtin_amdgcn_tanhf)
  return __builtin_amdgcn_tanhf(x);
#elif __has_builtin(__builtin_amdgcn_tanh_f32)
  return __builtin_amdgcn_tanh_f32(x);
#else
  float y = __builtin_amdgcn_fmed3f(x, -10.f, 10.f);
  float e = __builtin_amdgcn_exp2f(y * 2.8853900817779268f);
  return (e - 1.f) * __builtin_amdgcn_rcpf(e + 1.f);
#endif
}

// lo16 <-> hi16 half-wave exchange. v_permlanex16_b32 with identity selects is
// pure VALU (no LDS round-trip / s_wait_dscnt on the recurrence critical path).
// Round-4 evidence: lowers to v_permlanex16_b32, ds count dropped to 0.
__device__ __forceinline__ uint32_t xswap16(uint32_t v) {
#if __has_builtin(__builtin_amdgcn_permlanex16)
  return (uint32_t)__builtin_amdgcn_permlanex16((int)v, (int)v,
                                                0x76543210, 0xfedcba98,
                                                true, false);
#else
  return (uint32_t)__shfl_xor((unsigned)v, 16, 32);
#endif
}

__device__ __forceinline__ v8f wmma16(const v16h& a, const v16h& b, const v8f& c) {
  // v_wmma_f32_16x16x32_f16: D = A(16x32) * B(32x16) + C(16x16)
  return __builtin_amdgcn_wmma_f32_16x16x32_f16(false, a, false, b, (short)0, c,
                                                false, false);
}

// ---- Fused 2-layer tanh RNN + FC, one wave per 16-batch tile ----------------
// B layout (wave32): lanes 0-15 hold K=0..7 in V0..3, K=16..23 in V4..7 (2 packed f16 per VGPR),
// lanes 16-31 hold K=8..15 / K=24..31.  A layout: lanes 0-15 rows, K0..7 in V0..3,
// K16..23 in V4..7; lanes 16-31 same rows, K8..15 / K24..31.
//
// B-operand garbage policy: A1 is identically zero for K=8..15 and K=20..31, A2/Af for
// K=18..23 and K=26..31, so B values in those K slots never contribute
// (0 x finite = 0).  All candidate garbage values are tanh outputs (|v|<=1)
// or packed x samples (finite), so the 0 x Inf -> NaN hazard cannot trigger.
// This lets B1/Bf be built with no lane masking and B2 with only 5 merges.
__global__ __launch_bounds__(256)
void rnn2_wmma_kernel(const float* __restrict__ x,
                      const float* __restrict__ Wih1, const float* __restrict__ Whh1,
                      const float* __restrict__ bih1, const float* __restrict__ bhh1,
                      const float* __restrict__ Wih2, const float* __restrict__ Whh2,
                      const float* __restrict__ bih2, const float* __restrict__ bhh2,
                      const float* __restrict__ Wfc,  const float* __restrict__ bfc,
                      float* __restrict__ out, int Bsz, int T)
{
  const int  lane   = threadIdx.x & 31;
  const int  wave   = blockIdx.x * (blockDim.x >> 5) + (threadIdx.x >> 5);
  const int  base_b = wave * 16;
  if (base_b >= Bsz) return;          // wave-uniform: EXEC stays all-ones below

  const int  l  = lane & 15;
  const bool hi = lane >= 16;
  const bool lo = !hi;

  // ---- Build f16 A operands once (layer1, layer2, fc) ----
  AB16 a1, a2, af;
  for (int i = 0; i < 8; ++i) { a1.u[i] = 0u; a2.u[i] = 0u; af.u[i] = 0u; }
  if (l < 10) {
    const float* wh1 = Whh1 + l * 10;
    const float* wi2 = Wih2 + l * 10;
    const float* wh2 = Whh2 + l * 10;
    if (lo) {
      a1.u[0] = pkf16(Wih1[l*2+0], Wih1[l*2+1]);  // K0,1   <- x
      a1.u[1] = pkf16(wh1[0], wh1[1]);            // K2,3   <- h1[0,1]
      a1.u[2] = pkf16(wh1[2], wh1[3]);            // K4,5
      a1.u[3] = pkf16(wh1[4], wh1[5]);            // K6,7
      a1.u[4] = pkf16(wh1[6], wh1[7]);            // K16,17 <- h1[6,7]
      a1.u[5] = pkf16(wh1[8], wh1[9]);            // K18,19 <- h1[8,9]
      a2.u[0] = pkf16(wi2[0], wi2[1]);            // K0,1   <- h1
      a2.u[1] = pkf16(wi2[2], wi2[3]);
      a2.u[2] = pkf16(wi2[4], wi2[5]);
      a2.u[3] = pkf16(wi2[6], wi2[7]);
      a2.u[4] = pkf16(wi2[8], wi2[9]);            // K16,17 <- h1[8,9]
    } else {
      a2.u[0] = pkf16(wh2[0], wh2[1]);            // K8,9   <- h2
      a2.u[1] = pkf16(wh2[2], wh2[3]);
      a2.u[2] = pkf16(wh2[4], wh2[5]);
      a2.u[3] = pkf16(wh2[6], wh2[7]);
      a2.u[4] = pkf16(wh2[8], wh2[9]);            // K24,25
    }
  }
  if (l < 4) {
    const float* wf = Wfc + l * 10;
    if (lo) {
      af.u[0] = pkf16(wf[0], wf[1]);
      af.u[1] = pkf16(wf[2], wf[3]);
      af.u[2] = pkf16(wf[4], wf[5]);
      af.u[3] = pkf16(wf[6], wf[7]);
    } else {
      af.u[0] = pkf16(wf[8], wf[9]);              // K8,9
    }
  }

  // ---- C operands = per-row bias broadcast (C/D layout: VGPR r = row r|r+8) ----
  C8 c1, c2, cf;
  for (int r = 0; r < 8; ++r) {
    int row = r + (hi ? 8 : 0);
    float b1v = 0.f, b2v = 0.f, bfv = 0.f;
    if (row < 10) { b1v = bih1[row] + bhh1[row]; b2v = bih2[row] + bhh2[row]; }
    if (lo && r < 4) bfv = bfc[r];
    c1.f[r] = b1v; c2.f[r] = b2v; cf.f[r] = bfv;
  }

  // Per-lane x stream: lane n owns batch base_b+(n&15); 128B lines amortize 16
  // steps and the whole tensor (64 MiB) fits in the 192 MB L2.
  const float2* xp = (const float2*)x + (size_t)(base_b + l) * T;

  uint32_t h1a=0,h1b=0,h1c=0,h1d=0,h1e=0;   // h1 f16 pairs (valid lanes 0-15)
  uint32_t h2a=0,h2b=0,h2c=0,h2d=0,h2e=0;   // h2 f16 pairs (valid lanes 16-31)
  uint32_t q0=0,q1=0,q2=0,q3=0,s0=0;        // last-step h2 packs (for FC)

  #pragma unroll 2
  for (int t = 0; t < T; ++t) {
    float2   xv  = xp[t];
    uint32_t xpk = pkf16(xv.x, xv.y);

    // ---- layer 1: h1 = tanh([Wih1|Whh1] @ [x;h1] + b1) ----
    // No lane masking: A1 is zero in hi lanes (K8..15, K24..31), hi-lane garbage is inert.
    AB16 b1;
    b1.u[0] = xpk;                    // K0,1  = x_t
    b1.u[1] = h1a;                    // K2..7,16..19 = h1_{t-1}
    b1.u[2] = h1b;
    b1.u[3] = h1c;
    b1.u[4] = h1d;
    b1.u[5] = h1e;
    b1.u[6] = 0u; b1.u[7] = 0u;
    C8 d1; d1.v = wmma16(a1.h, b1.h, c1.v);

    float t0=th(d1.f[0]), t1=th(d1.f[1]), t2=th(d1.f[2]), t3=th(d1.f[3]);
    float t4=th(d1.f[4]), t5=th(d1.f[5]), t6=th(d1.f[6]), t7=th(d1.f[7]);
    uint32_t p0=pkf16(t0,t1), p1=pkf16(t2,t3), p2=pkf16(t4,t5), p3=pkf16(t6,t7);
    uint32_t p89 = xswap16(p0);       // rows 8,9 -> lanes 0-15 (VALU crosslane)
    h1a=p0; h1b=p1; h1c=p2; h1d=p3; h1e=p89;

    // ---- layer 2: h2 = tanh([Wih2|Whh2] @ [h1;h2] + b2) ----
    AB16 b2;                          // lo half: h1_t (K0..7,16,17); hi half: h2_{t-1} (K8..15,24,25)
    b2.u[0] = lo ? p0  : h2a;
    b2.u[1] = lo ? p1  : h2b;
    b2.u[2] = lo ? p2  : h2c;
    b2.u[3] = lo ? p3  : h2d;
    b2.u[4] = lo ? p89 : h2e;
    b2.u[5] = 0u; b2.u[6] = 0u; b2.u[7] = 0u;   // K18..23 / K26..31: A2 zero
    C8 d2; d2.v = wmma16(a2.h, b2.h, c2.v);

    float u0=th(d2.f[0]), u1=th(d2.f[1]), u2=th(d2.f[2]), u3=th(d2.f[3]);
    float u4=th(d2.f[4]), u5=th(d2.f[5]), u6=th(d2.f[6]), u7=th(d2.f[7]);
    q0=pkf16(u0,u1); q1=pkf16(u2,u3); q2=pkf16(u4,u5); q3=pkf16(u6,u7);
    s0          = xswap16(q0);        // rows 0,1 -> hi lanes; rows 8,9 -> lo lanes
    uint32_t s1 = xswap16(q1);
    uint32_t s2 = xswap16(q2);
    uint32_t s3 = xswap16(q3);
    h2a=s0; h2b=s1; h2c=s2; h2d=s3; h2e=q0;     // hi lanes carry h2 for next step
  }

  // ---- FC: out = Wfc @ h2_T + bfc (one more WMMA) ----
  // Af is zero for K>=10, so hi-lane / tail garbage in Bf is inert.
  AB16 bf;
  bf.u[0] = q0;                       // K0..7 = h2[0..7]   (lanes 0-15)
  bf.u[1] = q1;
  bf.u[2] = q2;
  bf.u[3] = q3;
  bf.u[4] = s0;                       // K16,17/K24,25: Af zero here; h2[8,9] enter as K8,9 via q0 hi lanes
  bf.u[5] = 0u; bf.u[6] = 0u; bf.u[7] = 0u;
  C8 df; df.v = wmma16(af.h, bf.h, cf.v);

  if (lo) {                           // rows 0..3 = the 4 outputs for batch base_b+l
    float4 o; o.x = df.f[0]; o.y = df.f[1]; o.z = df.f[2]; o.w = df.f[3];
    ((float4*)out)[base_b + l] = o;   // global_store_b128, fully coalesced
  }
}

extern "C" void kernel_launch(void* const* d_in, const int* in_sizes, int n_in,
                              void* d_out, int out_size, void* d_ws, size_t ws_size,
                              hipStream_t stream) {
  const float* x    = (const float*)d_in[0];
  const float* Wih1 = (const float*)d_in[1];
  const float* Whh1 = (const float*)d_in[2];
  const float* bih1 = (const float*)d_in[3];
  const float* bhh1 = (const float*)d_in[4];
  const float* Wih2 = (const float*)d_in[5];
  const float* Whh2 = (const float*)d_in[6];
  const float* bih2 = (const float*)d_in[7];
  const float* bhh2 = (const float*)d_in[8];
  const float* Wfc  = (const float*)d_in[9];
  const float* bfc  = (const float*)d_in[10];
  float* out = (float*)d_out;

  const int T   = 2048;
  const int Bsz = in_sizes[0] / (T * 2);          // 4096
  const int tiles = (Bsz + 15) / 16;              // 256 waves, 16 batch rows each
  const int waves_per_block = 8;                  // 256 threads = 8 wave32
  const int blocks = (tiles + waves_per_block - 1) / waves_per_block;

  rnn2_wmma_kernel<<<blocks, waves_per_block * 32, 0, stream>>>(
      x, Wih1, Whh1, bih1, bhh1, Wih2, Whh2, bih2, bhh2, Wfc, bfc, out, Bsz, T);
}
